// ContrastiveLoss_69561290326311
// MI455X (gfx1250) — compile-verified, matrix operand order checked by reading, not run on hardware
//
#include <hip/hip_runtime.h>
#include <hip/hip_bf16.h>
#include <cstdint>

typedef __attribute__((ext_vector_type(16))) _Float16 v16h;
typedef __attribute__((ext_vector_type(8)))  _Float16 v8h;
typedef __attribute__((ext_vector_type(8)))  float    v8f;

#define NROWS 8192
#define BHALF 4096
#define DIM   512
#define TAU   0.07f

// LDS tile geometry: 128 rows x 32 K-halfs per chunk, padded to 40 halfs/row
// (80B row stride -> ds bank = 20*r mod 64: conflict-free b128 fragment reads).
#define LDS_STRIDE 40
#define LDS_BUF_HALFS (128 * LDS_STRIDE)   // 5120 halfs = 10240 B per buffer

// ---------------------------------------------------------------------------
// CDNA5 async copy: GLOBAL_LOAD_ASYNC_TO_LDS_B128 (ASYNCcnt-tracked).
// vdst VGPR = LDS byte address (low 32 bits of a generic shared pointer are
// the LDS offset per the aperture mapping), vaddr = 64-bit global address.
// ---------------------------------------------------------------------------
__device__ __forceinline__ void async_copy_b128(uint32_t lds_addr, const void* g) {
  asm volatile("global_load_async_to_lds_b128 %0, %1, off"
               :: "v"(lds_addr), "v"((uint64_t)(uintptr_t)g)
               : "memory");
}
__device__ __forceinline__ void wait_async_le4() {
  asm volatile("s_wait_asynccnt 0x4" ::: "memory");
}
__device__ __forceinline__ void wait_async_0() {
  asm volatile("s_wait_asynccnt 0x0" ::: "memory");
}

// ---------------------------------------------------------------------------
// Fragment from LDS tile: 16-bit A-matrix 16x32 layout (ISA 7.12.2).
// lanes 0-15 : K = 0..7 (v0-3), 16..23 (v4-7); lanes 16-31: +8.
// B = R^T so B fragments use identical addressing with row = column.
// ---------------------------------------------------------------------------
__device__ __forceinline__ v16h lds_frag(const _Float16* __restrict__ buf,
                                         int row, int half) {
  const _Float16* p = buf + row * LDS_STRIDE + half * 8;
  v8h lo = *(const v8h*)(p);
  v8h hi = *(const v8h*)(p + 16);
  return __builtin_shufflevector(lo, hi, 0, 1, 2, 3, 4, 5, 6, 7,
                                 8, 9, 10, 11, 12, 13, 14, 15);
}

__device__ __forceinline__ v8f wmma_f16(v16h a, v16h b, v8f c) {
  return __builtin_amdgcn_wmma_f32_16x16x32_f16(
      false, a, false, b, (short)0, c, false, false);
}

// ---------------------------------------------------------------------------
// Kernel 1: row-normalize reps = [zi; zj] -> f16 matrix rnh [8192 x 512]
// ---------------------------------------------------------------------------
__global__ void normalize_kernel(const float* __restrict__ zi,
                                 const float* __restrict__ zj,
                                 _Float16* __restrict__ rnh) {
  const int row = blockIdx.x;
  const int t   = threadIdx.x;                // 256 threads
  const float* src = (row < BHALF) ? (zi + (size_t)row * DIM)
                                   : (zj + (size_t)(row - BHALF) * DIM);
  float x0 = src[t];
  float x1 = src[t + 256];
  float ss = x0 * x0 + x1 * x1;
#pragma unroll
  for (int m = 16; m > 0; m >>= 1) ss += __shfl_xor(ss, m, 32);

  __shared__ float smem[8];
  const int wave = t >> 5, lane = t & 31;
  if (lane == 0) smem[wave] = ss;
  __syncthreads();
  if (t == 0) {
    float s = 0.f;
#pragma unroll
    for (int i = 0; i < 8; ++i) s += smem[i];
    smem[0] = s;
  }
  __syncthreads();
  const float inv = 1.0f / fmaxf(sqrtf(smem[0]), 1e-8f);
  _Float16* dst = rnh + ((size_t)row << 9);
  dst[t]       = (_Float16)(x0 * inv);
  dst[t + 256] = (_Float16)(x1 * inv);
}

// ---------------------------------------------------------------------------
// Kernel 2: exact fp32 positives: pos[k] = cos(zi_k, zj_k)
// ---------------------------------------------------------------------------
__global__ void pos_kernel(const float* __restrict__ zi,
                           const float* __restrict__ zj,
                           float* __restrict__ pos) {
  const int k = blockIdx.x;
  const int t = threadIdx.x;                  // 256 threads
  const float* a = zi + (size_t)k * DIM;
  const float* b = zj + (size_t)k * DIM;
  float a0 = a[t], a1 = a[t + 256];
  float b0 = b[t], b1 = b[t + 256];
  float d  = a0 * b0 + a1 * b1;
  float sa = a0 * a0 + a1 * a1;
  float sb = b0 * b0 + b1 * b1;
#pragma unroll
  for (int m = 16; m > 0; m >>= 1) {
    d  += __shfl_xor(d,  m, 32);
    sa += __shfl_xor(sa, m, 32);
    sb += __shfl_xor(sb, m, 32);
  }
  __shared__ float sd[8], ssa[8], ssb[8];
  const int wave = t >> 5, lane = t & 31;
  if (lane == 0) { sd[wave] = d; ssa[wave] = sa; ssb[wave] = sb; }
  __syncthreads();
  if (t == 0) {
    float td = 0.f, ta = 0.f, tb = 0.f;
#pragma unroll
    for (int i = 0; i < 8; ++i) { td += sd[i]; ta += ssa[i]; tb += ssb[i]; }
    pos[k] = td / (fmaxf(sqrtf(ta), 1e-8f) * fmaxf(sqrtf(tb), 1e-8f));
  }
}

// ---------------------------------------------------------------------------
// Kernel 3: WMMA GEMM (sim = R * R^T) with async-LDS double buffering, fused
// exp(sim/tau) row-sums. 128x128 tile / 256-thread block (8 waves), each wave
// a 32x64 sub-tile = 2x4 fragments, K in 16 chunks of 32.
// Deterministic: partial[(bx*2+wc)*8192 + row] written by exactly one lane.
// ---------------------------------------------------------------------------
__global__ void __launch_bounds__(256)
gemm_exp_kernel(const _Float16* __restrict__ rnh,
                float* __restrict__ partial, float inv_tau) {
  __shared__ _Float16 lds[4 * LDS_BUF_HALFS];   // A[2] then B[2], 40 KB

  const int tid  = threadIdx.x;
  const int wave = tid >> 5;
  const int lane = tid & 31;
  const int half = lane >> 4;
  const int l16  = lane & 15;
  const int wr   = wave >> 1;        // 0..3 -> 32-row band
  const int wc   = wave & 1;         // 0..1 -> 64-col band

  const int rowBase0 = blockIdx.y * 128;
  const int colBase0 = blockIdx.x * 128;
  const uint32_t lbase = (uint32_t)(uintptr_t)lds;

  // Per-thread copy slots: elements e in [0,512): row = e>>2, 16B quarter q = e&3.
  const int eA0 = tid, eA1 = tid + 256;

  // Issue one K-chunk (A then B) into buffer `buf` (4 async b128 per thread).
  auto issue = [&](int kb, int buf) {
#pragma unroll
    for (int i = 0; i < 2; ++i) {
      const int e = (i == 0) ? eA0 : eA1;
      const int row = e >> 2, q = e & 3;
      const _Float16* g = rnh + ((size_t)(rowBase0 + row) << 9) + kb + q * 8;
      async_copy_b128(lbase + (uint32_t)(buf * 10240 + row * 80 + q * 16), g);
    }
#pragma unroll
    for (int i = 0; i < 2; ++i) {
      const int e = (i == 0) ? eA0 : eA1;
      const int row = e >> 2, q = e & 3;
      const _Float16* g = rnh + ((size_t)(colBase0 + row) << 9) + kb + q * 8;
      async_copy_b128(lbase + (uint32_t)(20480 + buf * 10240 + row * 80 + q * 16), g);
    }
  };

  v8f c[2][4] = {};

  issue(0, 0);
  for (int kc = 0; kc < 16; ++kc) {
    const int cur = kc & 1;
    if (kc < 15) {
      issue((kc + 1) * 32, cur ^ 1);
      wait_async_le4();          // oldest 4 (chunk kc) complete, in-order
    } else {
      wait_async_0();
    }
    __syncthreads();             // chunk kc visible to all waves

    const _Float16* At = lds + cur * LDS_BUF_HALFS;
    const _Float16* Bt = lds + 2 * LDS_BUF_HALFS + cur * LDS_BUF_HALFS;

    v16h a0 = lds_frag(At, wr * 32 + l16,      half);
    v16h a1 = lds_frag(At, wr * 32 + 16 + l16, half);
    v16h b0 = lds_frag(Bt, wc * 64 + l16,      half);
    v16h b1 = lds_frag(Bt, wc * 64 + 16 + l16, half);
    v16h b2 = lds_frag(Bt, wc * 64 + 32 + l16, half);
    v16h b3 = lds_frag(Bt, wc * 64 + 48 + l16, half);

    c[0][0] = wmma_f16(a0, b0, c[0][0]);
    c[0][1] = wmma_f16(a0, b1, c[0][1]);
    c[0][2] = wmma_f16(a0, b2, c[0][2]);
    c[0][3] = wmma_f16(a0, b3, c[0][3]);
    c[1][0] = wmma_f16(a1, b0, c[1][0]);
    c[1][1] = wmma_f16(a1, b1, c[1][1]);
    c[1][2] = wmma_f16(a1, b2, c[1][2]);
    c[1][3] = wmma_f16(a1, b3, c[1][3]);

    __syncthreads();             // reads done before buffer cur is re-filled
  }

  // Epilogue: C layout (VGPR r, lane) -> row = r + half*8, col = l16.
#pragma unroll
  for (int fr = 0; fr < 2; ++fr) {
#pragma unroll
    for (int r = 0; r < 8; ++r) {
      const int g_r = rowBase0 + wr * 32 + fr * 16 + r + half * 8;
      float v = 0.f;
#pragma unroll
      for (int fc = 0; fc < 4; ++fc) {
        const int g_c = colBase0 + wc * 64 + fc * 16 + l16;
        const float e = __expf(c[fr][fc][r] * inv_tau);
        v += (g_r == g_c) ? 0.f : e;
      }
#pragma unroll
      for (int m = 8; m > 0; m >>= 1) v += __shfl_xor(v, m, 32);
      if (l16 == 0)
        partial[(size_t)(blockIdx.x * 2 + wc) * NROWS + g_r] = v;
    }
  }
}

// ---------------------------------------------------------------------------
// Kernel 4: per-row deterministic reduction of 128 partials -> log(denom)
// ---------------------------------------------------------------------------
__global__ void rowsum_log_kernel(const float* __restrict__ partial,
                                  float* __restrict__ dlog) {
  const int row = blockIdx.x * 256 + threadIdx.x;   // 32 blocks x 256
  float s = 0.f;
  for (int j = 0; j < 128; ++j) s += partial[(size_t)j * NROWS + row];
  dlog[row] = logf(s);
}

// ---------------------------------------------------------------------------
// Kernel 5: loss = ( sum_i log(denom_i) - (2/tau) * sum_k pos_k ) / 8192
// ---------------------------------------------------------------------------
__global__ void finalize_kernel(const float* __restrict__ dlog,
                                const float* __restrict__ pos,
                                float* __restrict__ out, float inv_tau) {
  const int t = threadIdx.x;                  // 1024 threads
  float s_log = 0.f, s_pos = 0.f;
  for (int i = t; i < NROWS; i += 1024) s_log += dlog[i];
  for (int i = t; i < BHALF; i += 1024) s_pos += pos[i];
#pragma unroll
  for (int m = 16; m > 0; m >>= 1) {
    s_log += __shfl_xor(s_log, m, 32);
    s_pos += __shfl_xor(s_pos, m, 32);
  }
  __shared__ float sl[32], sp[32];
  const int wave = t >> 5, lane = t & 31;
  if (lane == 0) { sl[wave] = s_log; sp[wave] = s_pos; }
  __syncthreads();
  if (t == 0) {
    float tl = 0.f, tp = 0.f;
#pragma unroll
    for (int i = 0; i < 32; ++i) { tl += sl[i]; tp += sp[i]; }
    out[0] = (tl - 2.0f * inv_tau * tp) / (float)NROWS;
  }
}

// ---------------------------------------------------------------------------
extern "C" void kernel_launch(void* const* d_in, const int* in_sizes, int n_in,
                              void* d_out, int out_size, void* d_ws, size_t ws_size,
                              hipStream_t stream) {
  (void)in_sizes; (void)n_in; (void)out_size; (void)ws_size;
  const float* zi = (const float*)d_in[0];
  const float* zj = (const float*)d_in[1];
  float* out = (float*)d_out;

  char* ws = (char*)d_ws;
  _Float16* rnh   = (_Float16*)ws;                                  // 8 MB
  float* partial  = (float*)(ws + (size_t)8 * 1024 * 1024);         // 4 MB
  float* pos      = (float*)(ws + (size_t)12 * 1024 * 1024);        // 16 KB
  float* dlog     = (float*)(ws + (size_t)12 * 1024 * 1024 + 16384);// 32 KB

  normalize_kernel<<<NROWS, 256, 0, stream>>>(zi, zj, rnh);
  pos_kernel<<<BHALF, 256, 0, stream>>>(zi, zj, pos);

  dim3 grid(NROWS / 128, NROWS / 128);   // 64 x 64 tiles of 128x128
  gemm_exp_kernel<<<grid, 256, 0, stream>>>(rnh, partial, 1.0f / TAU);

  rowsum_log_kernel<<<NROWS / 256, 256, 0, stream>>>(partial, dlog);
  finalize_kernel<<<1, 1024, 0, stream>>>(dlog, pos, out, 1.0f / TAU);
}
